// SimpleMultiHeadAttention_30021821399310
// MI455X (gfx1250) — compile-verified, matrix operand order checked by reading, not run on hardware
//
#include <hip/hip_runtime.h>
#include <hip/hip_bf16.h>

typedef __bf16 bf16_t;
typedef __attribute__((ext_vector_type(16))) __bf16 v16bf;
typedef __attribute__((ext_vector_type(8)))  __bf16 v8bf;
typedef __attribute__((ext_vector_type(4)))  __bf16 v4bf;
typedef __attribute__((ext_vector_type(8)))  float  v8f;

constexpr int NB   = 4;      // batch
constexpr int LSEQ = 2048;   // sequence length
constexpr int EMB  = 1024;   // embed dim
constexpr int NH   = 16;     // heads
constexpr int DH   = 64;     // head dim
constexpr int NT   = NB * LSEQ;  // 8192 total tokens

// ---------------- helpers ----------------

__device__ __forceinline__ bf16_t f2bf(float f) {
  unsigned u = __builtin_bit_cast(unsigned, f);
  unsigned r = u + 0x7FFFu + ((u >> 16) & 1u);   // round to nearest even
  unsigned short h = (unsigned short)(r >> 16);
  return __builtin_bit_cast(bf16_t, h);
}

__device__ __forceinline__ v8f wmma_bf16(v16bf a, v16bf b, v8f c) {
  // D(f32 16x16) = A(bf16 16x32) * B(bf16 32x16) + C
  return __builtin_amdgcn_wmma_f32_16x16x32_bf16(
      false, a, false, b, (short)0, c, false, false);
}

// A-fragment (16x32, row-major rows with stride `stride`) or B-fragment when
// memory holds B columns contiguously (N x K row-major == K-major columns).
// Per ISA 16-bit A layout: lane l -> row (l&15); k = kbase + e (e<8) or
// kbase + 16 + (e-8), kbase = (l>>4)*8.  Two 16B loads per lane.
__device__ __forceinline__ v16bf load_frag(const bf16_t* base, int stride) {
  int l = threadIdx.x & 31;
  const bf16_t* p = base + (size_t)(l & 15) * stride + ((l >> 4) << 3);
  v8bf lo = *(const v8bf*)p;
  v8bf hi = *(const v8bf*)(p + 16);
  return __builtin_shufflevector(lo, hi, 0,1,2,3,4,5,6,7,8,9,10,11,12,13,14,15);
}

// A-fragment gathered from attention output stored (nb, h, l, d) when the
// logical matrix is (nb*L + l, h*64 + d).  Runs of 8 k's never cross a head.
__device__ __forceinline__ v16bf load_frag_attn(const bf16_t* O, int rowBase, int kt) {
  int l   = threadIdx.x & 31;
  int m   = rowBase + (l & 15);
  int nb  = m >> 11;               // /LSEQ
  int tok = m & (LSEQ - 1);
  int kb  = kt + ((l >> 4) << 3);
  int h0 = kb >> 6,        d0 = kb & 63;
  int h1 = (kb + 16) >> 6, d1 = (kb + 16) & 63;
  v8bf lo = *(const v8bf*)(O + (size_t)((nb * NH + h0) * LSEQ + tok) * DH + d0);
  v8bf hi = *(const v8bf*)(O + (size_t)((nb * NH + h1) * LSEQ + tok) * DH + d1);
  return __builtin_shufflevector(lo, hi, 0,1,2,3,4,5,6,7,8,9,10,11,12,13,14,15);
}

// ---------------- fp32 -> bf16 convert ----------------

__global__ void cvt_kernel(const float* __restrict__ src, bf16_t* __restrict__ dst, int n4) {
  int i = blockIdx.x * blockDim.x + threadIdx.x;
  if (i < n4) {
    float4 f = ((const float4*)src)[i];
    v4bf o;
    o[0] = f2bf(f.x); o[1] = f2bf(f.y); o[2] = f2bf(f.z); o[3] = f2bf(f.w);
    ((v4bf*)dst)[i] = o;
  }
}

// ---------------- GEMM: C = X(M=NT x K=EMB) * W^T (W is EMB x EMB) --------
// MODE 0: store bf16 at out[((nb*NH+h)*LSEQ+tok)*DH + d]         (Q, K)
// MODE 1: store bf16 at out[((nb*NH+h)*DH + d)*LSEQ + tok]       (V^T)
// MODE 2: A gathered from attn layout; store f32 at out[m*EMB+n] (final)

template <int MODE>
__global__ __launch_bounds__(128) void gemm128(const bf16_t* __restrict__ X,
                                               const bf16_t* __restrict__ W,
                                               void* __restrict__ outv) {
  const int wave  = threadIdx.x >> 5;
  const int lane  = threadIdx.x & 31;
  const int mBase = blockIdx.x * 128 + (wave >> 1) * 64;
  const int nBase = blockIdx.y * 128 + (wave & 1) * 64;

  v8f acc[4][4] = {};
  for (int kt = 0; kt < EMB; kt += 32) {
    v16bf a[4], b[4];
#pragma unroll
    for (int i = 0; i < 4; i++) {
      if constexpr (MODE == 2)
        a[i] = load_frag_attn(X, mBase + i * 16, kt);
      else
        a[i] = load_frag(X + (size_t)(mBase + i * 16) * EMB + kt, EMB);
    }
#pragma unroll
    for (int j = 0; j < 4; j++)
      b[j] = load_frag(W + (size_t)(nBase + j * 16) * EMB + kt, EMB);

    // Prefetch the streaming A tiles two k-steps ahead (weights are L2-hot).
    if constexpr (MODE != 2) {
      if (kt + 64 < EMB) {
#pragma unroll
        for (int i = 0; i < 4; i++)
          __builtin_prefetch(X + (size_t)(mBase + i * 16 + (lane & 15)) * EMB + kt + 64, 0, 3);
      }
    }

#pragma unroll
    for (int i = 0; i < 4; i++)
#pragma unroll
      for (int j = 0; j < 4; j++)
        acc[i][j] = wmma_bf16(a[i], b[j], acc[i][j]);
  }

  const int nOff = lane & 15;
  const int mOff = (lane >> 4) * 8;
#pragma unroll
  for (int i = 0; i < 4; i++)
#pragma unroll
    for (int j = 0; j < 4; j++)
#pragma unroll
      for (int r = 0; r < 8; r++) {
        int m = mBase + i * 16 + mOff + r;
        int n = nBase + j * 16 + nOff;
        float v = acc[i][j][r];
        if constexpr (MODE == 2) {
          ((float*)outv)[(size_t)m * EMB + n] = v;
        } else {
          int nb = m >> 11, tok = m & (LSEQ - 1);
          int h = n >> 6, d = n & 63;
          bf16_t* o = (bf16_t*)outv;
          if constexpr (MODE == 0)
            o[(((size_t)(nb * NH + h) * LSEQ + tok) << 6) + d] = f2bf(v);
          else
            o[(((size_t)(nb * NH + h) * DH + d) << 11) + tok] = f2bf(v);
        }
      }
}

// ---------------- causal flash attention ----------------
// Q,K: (nb,h,l,d) bf16;  Vt: (nb,h,d,l) bf16;  O: (nb,h,l,d) bf16
// One wave owns 16 q-rows, kv scanned in steps of 64.  All K fragments are
// loaded into live arrays before the S WMMAs (clause-grouped, staged waits);
// V fragments are issued before the softmax block so their latency hides
// under the VALU work.  Row sums come from P x ones via WMMA; only the row
// max needs cross-lane shuffles.

__global__ __launch_bounds__(128) void flash_attn(const bf16_t* __restrict__ Q,
                                                  const bf16_t* __restrict__ K,
                                                  const bf16_t* __restrict__ Vt,
                                                  bf16_t* __restrict__ O) {
  __shared__ __align__(16) bf16_t sc[4][16 * 64];   // per-wave P transpose
  const int wave = threadIdx.x >> 5;
  const int lane = threadIdx.x & 31;
  const int nOff = lane & 15;
  const int mOff = (lane >> 4) * 8;

  const size_t head = blockIdx.y;                   // nb*NH + h
  const bf16_t* Qh = Q  + head * (size_t)LSEQ * DH;
  const bf16_t* Kh = K  + head * (size_t)LSEQ * DH;
  const bf16_t* Vh = Vt + head * (size_t)DH * LSEQ;
  bf16_t*       Oh = O  + head * (size_t)LSEQ * DH;

  const int qBase = blockIdx.x * 64 + wave * 16;
  v16bf q0 = load_frag(Qh + (size_t)qBase * DH, DH);        // d = 0..31
  v16bf q1 = load_frag(Qh + (size_t)qBase * DH + 32, DH);   // d = 32..63

  // all-ones B matrix: row sums of P via WMMA
  const bf16_t oneb = __builtin_bit_cast(bf16_t, (unsigned short)0x3F80);
  v16bf onesf;
#pragma unroll
  for (int e = 0; e < 16; e++) onesf[e] = oneb;

  v8f o0 = {}, o1 = {}, o2 = {}, o3 = {};
  float mrow[8], lrow[8], alpha[8];
#pragma unroll
  for (int r = 0; r < 8; r++) { mrow[r] = -1e30f; lrow[r] = 0.f; }

  bf16_t* my = sc[wave];
  const int jEnd = qBase + 15;          // per-wave causal end
  for (int j = 0; j <= jEnd; j += 64) {
    // ---- load all K fragments, then S = Q K^T (4 x 16x16 tiles) ----
    v16bf ka[4], kb[4];
#pragma unroll
    for (int c = 0; c < 4; c++) {
      ka[c] = load_frag(Kh + (size_t)(j + 16 * c) * DH, DH);
      kb[c] = load_frag(Kh + (size_t)(j + 16 * c) * DH + 32, DH);
    }
    v8f s[4];
#pragma unroll
    for (int c = 0; c < 4; c++) {
      v8f t = {};
      t = wmma_bf16(q0, ka[c], t);
      t = wmma_bf16(q1, kb[c], t);
      s[c] = t;
    }

    // ---- issue V fragment loads early (independent of softmax) ----
    v16bf va[4], vb[4];
#pragma unroll
    for (int f = 0; f < 4; f++) {
      va[f] = load_frag(Vh + (size_t)(f * 16) * LSEQ + j, LSEQ);
      vb[f] = load_frag(Vh + (size_t)(f * 16) * LSEQ + j + 32, LSEQ);
    }

    // ---- softmax: mask/scale, row max (shuffles), exp -> LDS ----
#pragma unroll
    for (int r = 0; r < 8; r++) {
      int qRow = qBase + mOff + r;
      float e[4];
#pragma unroll
      for (int c = 0; c < 4; c++) {
        float x = s[c][r] * 0.125f;
        if (j + 16 * c + nOff > qRow) x = -1e30f;
        e[c] = x;
      }
      float mx = fmaxf(fmaxf(e[0], e[1]), fmaxf(e[2], e[3]));
      mx = fmaxf(mx, __shfl_xor(mx, 1, 32));
      mx = fmaxf(mx, __shfl_xor(mx, 2, 32));
      mx = fmaxf(mx, __shfl_xor(mx, 4, 32));
      mx = fmaxf(mx, __shfl_xor(mx, 8, 32));
      float mn = fmaxf(mrow[r], mx);
      alpha[r] = __expf(mrow[r] - mn);
      mrow[r] = mn;
#pragma unroll
      for (int c = 0; c < 4; c++)
        my[(mOff + r) * 64 + c * 16 + nOff] = f2bf(__expf(e[c] - mn));
    }

    // ---- rescale O by alpha while the DS stores drain ----
#pragma unroll
    for (int r = 0; r < 8; r++) {
      o0[r] *= alpha[r]; o1[r] *= alpha[r];
      o2[r] *= alpha[r]; o3[r] *= alpha[r];
    }

    // Same-wave DS ops execute in order; pin compiler ordering only.
    __builtin_amdgcn_wave_barrier();
    asm volatile("" ::: "memory");
    v16bf p0 = load_frag(my, 64);        // kv offsets  0..31
    v16bf p1 = load_frag(my + 32, 64);   // kv offsets 32..63

    // ---- row sums via P x ones (replaces 32 bpermutes/iter) ----
    v8f ssum = {};
    ssum = wmma_bf16(p0, onesf, ssum);
    ssum = wmma_bf16(p1, onesf, ssum);
#pragma unroll
    for (int r = 0; r < 8; r++)
      lrow[r] = lrow[r] * alpha[r] + ssum[r];

    // ---- O += P * V  (V^T layout gives contiguous B loads) ----
    o0 = wmma_bf16(p0, va[0], o0); o0 = wmma_bf16(p1, vb[0], o0);
    o1 = wmma_bf16(p0, va[1], o1); o1 = wmma_bf16(p1, vb[1], o1);
    o2 = wmma_bf16(p0, va[2], o2); o2 = wmma_bf16(p1, vb[2], o2);
    o3 = wmma_bf16(p0, va[3], o3); o3 = wmma_bf16(p1, vb[3], o3);
  }

  // ---- normalize and store (bf16, (nb,h,l,d)) ----
#pragma unroll
  for (int r = 0; r < 8; r++) {
    float inv = 1.0f / lrow[r];
    int row = qBase + mOff + r;
    Oh[(size_t)row * DH + 0  + nOff] = f2bf(o0[r] * inv);
    Oh[(size_t)row * DH + 16 + nOff] = f2bf(o1[r] * inv);
    Oh[(size_t)row * DH + 32 + nOff] = f2bf(o2[r] * inv);
    Oh[(size_t)row * DH + 48 + nOff] = f2bf(o3[r] * inv);
  }
}

// ---------------- host side ----------------

extern "C" void kernel_launch(void* const* d_in, const int* in_sizes, int n_in,
                              void* d_out, int out_size, void* d_ws, size_t ws_size,
                              hipStream_t stream) {
  (void)in_sizes; (void)n_in; (void)out_size; (void)ws_size;
  const float* q_in = (const float*)d_in[0];
  const float* k_in = (const float*)d_in[1];
  const float* v_in = (const float*)d_in[2];
  const float* wq   = (const float*)d_in[3];
  const float* wk   = (const float*)d_in[4];
  const float* wv   = (const float*)d_in[5];
  const float* wo   = (const float*)d_in[6];
  // d_in[7] is the causal mask; applied analytically in flash_attn.

  const size_t TOKE = (size_t)NT * EMB;   // 8.39M elements
  const size_t WSZ  = (size_t)EMB * EMB;  // 1.05M elements
  char* ws = (char*)d_ws;
  bf16_t* qx  = (bf16_t*)ws; ws += TOKE * 2;
  bf16_t* kx  = (bf16_t*)ws; ws += TOKE * 2;
  bf16_t* vx  = (bf16_t*)ws; ws += TOKE * 2;
  bf16_t* wqb = (bf16_t*)ws; ws += WSZ * 2;
  bf16_t* wkb = (bf16_t*)ws; ws += WSZ * 2;
  bf16_t* wvb = (bf16_t*)ws; ws += WSZ * 2;
  bf16_t* wob = (bf16_t*)ws; ws += WSZ * 2;
  bf16_t* Qb  = (bf16_t*)ws; ws += TOKE * 2;
  bf16_t* Kb  = (bf16_t*)ws; ws += TOKE * 2;
  bf16_t* Vtb = (bf16_t*)ws; ws += TOKE * 2;
  bf16_t* Ob  = qx;   // qx is dead after the Q projection; reuse for attn out

  const int n4t = (int)(TOKE / 4);
  const int n4w = (int)(WSZ / 4);
  cvt_kernel<<<(n4t + 255) / 256, 256, 0, stream>>>(q_in, qx, n4t);
  cvt_kernel<<<(n4t + 255) / 256, 256, 0, stream>>>(k_in, kx, n4t);
  cvt_kernel<<<(n4t + 255) / 256, 256, 0, stream>>>(v_in, vx, n4t);
  cvt_kernel<<<(n4w + 255) / 256, 256, 0, stream>>>(wq, wqb, n4w);
  cvt_kernel<<<(n4w + 255) / 256, 256, 0, stream>>>(wk, wkb, n4w);
  cvt_kernel<<<(n4w + 255) / 256, 256, 0, stream>>>(wv, wvb, n4w);
  cvt_kernel<<<(n4w + 255) / 256, 256, 0, stream>>>(wo, wob, n4w);

  dim3 gg(NT / 128, EMB / 128);   // (64, 8)
  gemm128<0><<<gg, 128, 0, stream>>>(qx, wqb, Qb);
  gemm128<0><<<gg, 128, 0, stream>>>(kx, wkb, Kb);
  gemm128<1><<<gg, 128, 0, stream>>>(vx, wvb, Vtb);

  flash_attn<<<dim3(LSEQ / 64, NB * NH), 128, 0, stream>>>(Qb, Kb, Vtb, Ob);

  gemm128<2><<<gg, 128, 0, stream>>>(Ob, wob, d_out);
}
